// DifferentialAttention_28913719837071
// MI455X (gfx1250) — matrix operand order repaired and checked by reading, not compiled, not run on hardware
//
#include <hip/hip_runtime.h>
#include <stdint.h>

// Problem constants (reference: B=4, N=1024, C=1024, H=16, HD=64)
#define B_   4
#define N_   1024
#define C_   1024
#define H_   16
#define HD_  64
#define LAMBDA_INIT 0.8f

typedef __attribute__((ext_vector_type(16))) __bf16 v16bf;
typedef __attribute__((ext_vector_type(8)))  float  v8f;

// ---- helpers --------------------------------------------------------------

static __device__ __forceinline__ __bf16 f2bf(float f) {
  union { float f; uint32_t u; } in; in.f = f;
  uint32_t r = (in.u + 0x7FFFu + ((in.u >> 16) & 1u)) >> 16;  // RNE
  union { uint16_t u; __bf16 b; } out; out.u = (uint16_t)r;
  return out.b;
}

// A-matrix (16x32 bf16) per-lane K index (ISA 7.12.2):
// lanes 0-15: K = {0..7,16..23}; lanes 16-31: K = {8..15,24..31}
static __device__ __forceinline__ int kmapA(int lane, int e) {
  int half = (lane & 16) ? 8 : 0;
  return (e < 8) ? (e + half) : (e + 8 + half);
}
// B-matrix (32x16 bf16) per-lane K index (same as A): lanes 0-15 K={0..7,16..23}; lanes 16-31 K={8..15,24..31}
static __device__ __forceinline__ int kmapB(int lane, int e) {
  int half = (lane & 16) ? 8 : 0;
  return (e < 8) ? (e + half) : (e + 8 + half);
}

// CDNA5 async global->LDS copy (ASYNCcnt-tracked)
static __device__ __forceinline__ void async_copy_b128(uint32_t lds_off, const void* gptr) {
  asm volatile("global_load_async_to_lds_b128 %0, %1, off"
               :: "v"(lds_off), "v"(gptr) : "memory");
}
static __device__ __forceinline__ void wait_async0() {
  asm volatile("s_wait_asynccnt 0" ::: "memory");
}

// ---- kernel: fp32 -> bf16 conversion --------------------------------------

__global__ void cvt_f32_bf16(const float* __restrict__ src, __bf16* __restrict__ dst, int n) {
  int i = blockIdx.x * blockDim.x + threadIdx.x;
  if (i < n) dst[i] = f2bf(src[i]);
}

// ---- kernel: lambda scalar -------------------------------------------------

__global__ void lam_kernel(const float* __restrict__ lq1, const float* __restrict__ lk1,
                           const float* __restrict__ lq2, const float* __restrict__ lk2,
                           float* __restrict__ lam) {
  __shared__ float red[64];
  int t = threadIdx.x;  // 64 threads
  red[t] = __expf(lq1[t] * lk1[t]) - __expf(lq2[t] * lk2[t]) + LAMBDA_INIT;
  __syncthreads();
  for (int s = 32; s > 0; s >>= 1) {
    if (t < s) red[t] += red[t + s];
    __syncthreads();
  }
  if (t == 0) *lam = red[0] * (1.0f / 64.0f);
}

// ---- kernel: QKV GEMM (4096x1024 @ 1024x3072) -----------------------------
// grid (64, 12), block 256 (8 waves in 2x4). Block tile 64(M) x 256(N),
// K-tiles of 32 double-buffered in LDS via async copies. Each wave holds
// a 2x4 grid of 16x16 fp32 accumulators -> 8 WMMAs per K-step.

__global__ void __launch_bounds__(256) qkv_gemm(const __bf16* __restrict__ xb,
                                                const __bf16* __restrict__ wq,
                                                __bf16* __restrict__ qb,
                                                __bf16* __restrict__ kb,
                                                __bf16* __restrict__ vb) {
  __shared__ alignas(16) __bf16 shA[2][64 * 32];   // 8 KB
  __shared__ alignas(16) __bf16 shB[2][256 * 32];  // 32 KB
  const int t = threadIdx.x;
  const int lane = t & 31, wave = t >> 5;
  const int wm = wave >> 2, wn = wave & 3;
  const int mbase = blockIdx.x * 64;
  const int nbase = blockIdx.y * 256;
  const uint32_t ldsA = (uint32_t)(uintptr_t)(void*)&shA[0][0];
  const uint32_t ldsB = (uint32_t)(uintptr_t)(void*)&shB[0][0];
  const int arow = t >> 2, aq = t & 3;  // A tile: 64 rows x 4 b128 packets

  auto stage = [&](int buf, int kt) {
    const char* ga = (const char*)(xb + (size_t)(mbase + arow) * 1024 + kt * 32 + aq * 8);
    async_copy_b128(ldsA + (uint32_t)(buf * 64 * 32 + arow * 32 + aq * 8) * 2u, ga);
    const char* gb = (const char*)(wq + (size_t)(nbase + t) * 1024 + kt * 32);
    #pragma unroll
    for (int q = 0; q < 4; q++)
      async_copy_b128(ldsB + (uint32_t)(buf * 256 * 32 + t * 32 + q * 8) * 2u, gb + q * 16);
  };

  v8f acc[2][4] = {};
  stage(0, 0);
  int buf = 0;
  for (int kt = 0; kt < 32; kt++) {
    wait_async0();
    __syncthreads();                 // staged K-tile visible; prior readers done
    if (kt + 1 < 32) stage(buf ^ 1, kt + 1);
    v16bf af[2], bfr[4];
    #pragma unroll
    for (int mi = 0; mi < 2; mi++)
      #pragma unroll
      for (int e = 0; e < 16; e++)
        af[mi][e] = shA[buf][(wm * 32 + mi * 16 + (lane & 15)) * 32 + kmapA(lane, e)];
    #pragma unroll
    for (int ni = 0; ni < 4; ni++)
      #pragma unroll
      for (int e = 0; e < 16; e++)
        bfr[ni][e] = shB[buf][(wn * 64 + ni * 16 + (lane & 15)) * 32 + kmapB(lane, e)];
    #pragma unroll
    for (int mi = 0; mi < 2; mi++)
      #pragma unroll
      for (int ni = 0; ni < 4; ni++)
        acc[mi][ni] = __builtin_amdgcn_wmma_f32_16x16x32_bf16(
            false, af[mi], false, bfr[ni], (short)0, acc[mi][ni], false, false);
    buf ^= 1;
  }

  // scatter into per-head q/k/v layout [B*H][N][HD]
  const int hi8 = (lane & 16) ? 8 : 0;
  #pragma unroll
  for (int mi = 0; mi < 2; mi++) {
    #pragma unroll
    for (int ni = 0; ni < 4; ni++) {
      const int ncol = nbase + wn * 64 + ni * 16 + (lane & 15);
      const int sect = ncol >> 10;          // 0=q 1=k 2=v (tile-uniform)
      const int cc = ncol & 1023;
      const int h = cc >> 6, d = cc & 63;
      __bf16* dst = (sect == 0) ? qb : ((sect == 1) ? kb : vb);
      #pragma unroll
      for (int r = 0; r < 8; r++) {
        int gm = mbase + wm * 32 + mi * 16 + r + hi8;
        int bidx = gm >> 10, n = gm & 1023;
        dst[(((size_t)bidx * H_ + h) * N_ + n) * HD_ + d] = f2bf(acc[mi][ni][r]);
      }
    }
  }
}

// ---- kernel: differential attention ---------------------------------------
// grid 4096 = B*H*(N/16); block 128 (4 waves). Score stripe (2x 16x1024 f32 =
// 128 KB) stays LDS-resident -- possible only with CDNA5's 320 KB WGP LDS;
// avoids ~1 GB of HBM traffic for the (B,H,N,N) score tensors.
// K and V tiles are async-staged coalesced into a shared 8 KB LDS buffer.

__global__ void __launch_bounds__(128) diff_attn(const __bf16* __restrict__ qbm,
                                                 const __bf16* __restrict__ kbm,
                                                 const __bf16* __restrict__ vbm,
                                                 const float* __restrict__ lamp,
                                                 __bf16* __restrict__ ob) {
  extern __shared__ float smemf[];
  float* S1 = smemf;                            // 16*1024 f32 (64 KB)
  float* S2 = smemf + 16 * 1024;                // 16*1024 f32 (64 KB)
  __bf16* Qt  = (__bf16*)(smemf + 32 * 1024);   // 16*64 bf16 (2 KB)
  __bf16* KVt = Qt + 16 * 64;                   // 64*64 bf16 (8 KB)
  __shared__ float red[128];
  __shared__ float rmax[16], rsum1[16], rsum2[16];

  const int t = threadIdx.x;
  const int bh = blockIdx.x >> 6;        // b*H + h
  const int rtile = blockIdx.x & 63;
  const int b = bh >> 4, h = bh & 15;
  const int lane = t & 31, wave = t >> 5;
  const size_t hbase = (size_t)bh * N_ * HD_;
  const uint32_t ldsQ  = (uint32_t)(uintptr_t)(void*)Qt;
  const uint32_t ldsKV = (uint32_t)(uintptr_t)(void*)KVt;

  // stage 16x64 Q tile (2 KB, contiguous) via async copy
  {
    const char* gsrc = (const char*)(qbm + hbase + (size_t)rtile * 16 * HD_);
    async_copy_b128(ldsQ + (uint32_t)t * 16u, gsrc + t * 16);
    wait_async0();
  }
  __syncthreads();

  const int row = lane & 15;
  const int hi8 = (lane & 16) ? 8 : 0;

  // q1/q2 A-fragments (head-dim halves 0..31 / 32..63)
  v16bf a1, a2;
  #pragma unroll
  for (int e = 0; e < 16; e++) {
    int kk = kmapA(lane, e);
    a1[e] = Qt[row * 64 + kk];
    a2[e] = Qt[row * 64 + 32 + kk];
  }

  // scores: 16 stages of 64 keys; K-tile staged coalesced, each wave one
  // 16-key chunk -> two WMMAs (q1k1, q2k2) per stage per wave
  for (int i = 0; i < 16; i++) {
    const int kb0 = i * 64;
    const char* gsrc = (const char*)(kbm + hbase + (size_t)kb0 * HD_);
    #pragma unroll
    for (int j = 0; j < 4; j++)
      async_copy_b128(ldsKV + (uint32_t)(t + j * 128) * 16u, gsrc + (t + j * 128) * 16);
    wait_async0();
    __syncthreads();
    v16bf b1, b2;
    #pragma unroll
    for (int e = 0; e < 16; e++) {
      int kk = kmapB(lane, e);
      b1[e] = KVt[(wave * 16 + (lane & 15)) * 64 + kk];
      b2[e] = KVt[(wave * 16 + (lane & 15)) * 64 + 32 + kk];
    }
    v8f c1 = {}, c2 = {};
    c1 = __builtin_amdgcn_wmma_f32_16x16x32_bf16(false, a1, false, b1, (short)0, c1, false, false);
    c2 = __builtin_amdgcn_wmma_f32_16x16x32_bf16(false, a2, false, b2, (short)0, c2, false, false);
    const int key0 = kb0 + wave * 16;
    #pragma unroll
    for (int r = 0; r < 8; r++) {
      int rr = r + hi8, col = key0 + (lane & 15);
      S1[rr * 1024 + col] = c1[r] * 0.125f;  // HD^-0.5
      S2[rr * 1024 + col] = c2[r] * 0.125f;
    }
    __syncthreads();  // KVt re-staged next iteration
  }

  // two in-LDS softmaxes, 8 threads per row
  const int srow = t >> 3, sub = t & 7;
  float m = -3.0e38f;
  for (int i = sub; i < 1024; i += 8) m = fmaxf(m, S1[srow * 1024 + i]);
  red[t] = m; __syncthreads();
  if (sub == 0) { float mm = red[t]; for (int j = 1; j < 8; j++) mm = fmaxf(mm, red[t + j]); rmax[srow] = mm; }
  __syncthreads();
  float s = 0.f, mrow = rmax[srow];
  for (int i = sub; i < 1024; i += 8) { float e = __expf(S1[srow * 1024 + i] - mrow); S1[srow * 1024 + i] = e; s += e; }
  red[t] = s; __syncthreads();
  if (sub == 0) { float ss = red[t]; for (int j = 1; j < 8; j++) ss += red[t + j]; rsum1[srow] = ss; }
  __syncthreads();

  m = -3.0e38f;
  for (int i = sub; i < 1024; i += 8) m = fmaxf(m, S2[srow * 1024 + i]);
  red[t] = m; __syncthreads();
  if (sub == 0) { float mm = red[t]; for (int j = 1; j < 8; j++) mm = fmaxf(mm, red[t + j]); rmax[srow] = mm; }
  __syncthreads();
  s = 0.f; mrow = rmax[srow];
  for (int i = sub; i < 1024; i += 8) { float e = __expf(S2[srow * 1024 + i] - mrow); S2[srow * 1024 + i] = e; s += e; }
  red[t] = s; __syncthreads();
  if (sub == 0) { float ss = red[t]; for (int j = 1; j < 8; j++) ss += red[t + j]; rsum2[srow] = ss; }
  __syncthreads();

  // combine: attn = softmax1 - lam * softmax2
  const float lam = *lamp;
  {
    float i1 = 1.0f / rsum1[srow];
    float i2 = lam / rsum2[srow];
    for (int i = sub; i < 1024; i += 8)
      S1[srow * 1024 + i] = S1[srow * 1024 + i] * i1 - S2[srow * 1024 + i] * i2;
  }
  __syncthreads();

  // attn @ V: 16 stages; 64x64 V-tile async-staged, each wave one 16-wide
  // HD chunk, two K=32 WMMAs per stage
  const int colg = wave * 16 + (lane & 15);
  v8f cacc = {};
  for (int i = 0; i < 16; i++) {
    const int k0 = i * 64;
    const char* gsrc = (const char*)(vbm + hbase + (size_t)k0 * HD_);
    #pragma unroll
    for (int j = 0; j < 4; j++)
      async_copy_b128(ldsKV + (uint32_t)(t + j * 128) * 16u, gsrc + (t + j * 128) * 16);
    wait_async0();
    __syncthreads();
    #pragma unroll
    for (int sstep = 0; sstep < 2; sstep++) {
      v16bf a, bb;
      #pragma unroll
      for (int e = 0; e < 16; e++) {
        int ka = kmapA(lane, e), kv = kmapB(lane, e);
        a[e]  = f2bf(S1[row * 1024 + k0 + sstep * 32 + ka]);
        bb[e] = KVt[(sstep * 32 + kv) * 64 + colg];
      }
      cacc = __builtin_amdgcn_wmma_f32_16x16x32_bf16(false, a, false, bb, (short)0, cacc, false, false);
    }
    __syncthreads();
  }
  #pragma unroll
  for (int r = 0; r < 8; r++) {
    int n = rtile * 16 + r + hi8;
    ob[((size_t)b * N_ + n) * C_ + h * HD_ + colg] = f2bf(cacc[r]);
  }
}

// ---- kernel: output projection + bias (4096x1024 @ 1024x1024) -------------
// grid (64, 4), block 256. Same 64x256 double-buffered tile as qkv_gemm.

__global__ void __launch_bounds__(256) proj_gemm(const __bf16* __restrict__ ab,
                                                 const __bf16* __restrict__ wp,
                                                 const float* __restrict__ bias,
                                                 float* __restrict__ out) {
  __shared__ alignas(16) __bf16 shA[2][64 * 32];
  __shared__ alignas(16) __bf16 shB[2][256 * 32];
  const int t = threadIdx.x;
  const int lane = t & 31, wave = t >> 5;
  const int wm = wave >> 2, wn = wave & 3;
  const int mbase = blockIdx.x * 64;
  const int nbase = blockIdx.y * 256;
  const uint32_t ldsA = (uint32_t)(uintptr_t)(void*)&shA[0][0];
  const uint32_t ldsB = (uint32_t)(uintptr_t)(void*)&shB[0][0];
  const int arow = t >> 2, aq = t & 3;

  auto stage = [&](int buf, int kt) {
    const char* ga = (const char*)(ab + (size_t)(mbase + arow) * 1024 + kt * 32 + aq * 8);
    async_copy_b128(ldsA + (uint32_t)(buf * 64 * 32 + arow * 32 + aq * 8) * 2u, ga);
    const char* gb = (const char*)(wp + (size_t)(nbase + t) * 1024 + kt * 32);
    #pragma unroll
    for (int q = 0; q < 4; q++)
      async_copy_b128(ldsB + (uint32_t)(buf * 256 * 32 + t * 32 + q * 8) * 2u, gb + q * 16);
  };

  v8f acc[2][4] = {};
  stage(0, 0);
  int buf = 0;
  for (int kt = 0; kt < 32; kt++) {
    wait_async0();
    __syncthreads();
    if (kt + 1 < 32) stage(buf ^ 1, kt + 1);
    v16bf af[2], bfr[4];
    #pragma unroll
    for (int mi = 0; mi < 2; mi++)
      #pragma unroll
      for (int e = 0; e < 16; e++)
        af[mi][e] = shA[buf][(wm * 32 + mi * 16 + (lane & 15)) * 32 + kmapA(lane, e)];
    #pragma unroll
    for (int ni = 0; ni < 4; ni++)
      #pragma unroll
      for (int e = 0; e < 16; e++)
        bfr[ni][e] = shB[buf][(wn * 64 + ni * 16 + (lane & 15)) * 32 + kmapB(lane, e)];
    #pragma unroll
    for (int mi = 0; mi < 2; mi++)
      #pragma unroll
      for (int ni = 0; ni < 4; ni++)
        acc[mi][ni] = __builtin_amdgcn_wmma_f32_16x16x32_bf16(
            false, af[mi], false, bfr[ni], (short)0, acc[mi][ni], false, false);
    buf ^= 1;
  }

  const int hi8 = (lane & 16) ? 8 : 0;
  #pragma unroll
  for (int mi = 0; mi < 2; mi++) {
    #pragma unroll
    for (int ni = 0; ni < 4; ni++) {
      const int ncol = nbase + wn * 64 + ni * 16 + (lane & 15);
      const float bv = bias[ncol];
      #pragma unroll
      for (int r = 0; r < 8; r++) {
        int gm = mbase + wm * 32 + mi * 16 + r + hi8;
        out[(size_t)gm * 1024 + ncol] = acc[mi][ni][r] + bv;
      }
    }
  }
}

// ---- launch ---------------------------------------------------------------

extern "C" void kernel_launch(void* const* d_in, const int* in_sizes, int n_in,
                              void* d_out, int out_size, void* d_ws, size_t ws_size,
                              hipStream_t stream) {
  (void)in_sizes; (void)n_in; (void)out_size; (void)ws_size;
  const float* x     = (const float*)d_in[0];
  const float* Wqkv  = (const float*)d_in[1];
  const float* Wproj = (const float*)d_in[2];
  const float* bproj = (const float*)d_in[3];
  const float* lq1   = (const float*)d_in[4];
  const float* lk1   = (const float*)d_in[5];
  const float* lq2   = (const float*)d_in[6];
  const float* lk2   = (const float*)d_in[7];

  char* ws = (char*)d_ws;
  __bf16* xb     = (__bf16*)(ws);              // 4M elems -> 8 MB
  __bf16* wqkvb  = (__bf16*)(ws + 8388608);    // 3M -> 6 MB
  __bf16* wprojb = (__bf16*)(ws + 14680064);   // 1M -> 2 MB
  __bf16* qb     = (__bf16*)(ws + 16777216);   // 8 MB
  __bf16* kb     = (__bf16*)(ws + 25165824);   // 8 MB
  __bf16* vb     = (__bf16*)(ws + 33554432);   // 8 MB
  __bf16* ob     = (__bf16*)(ws + 41943040);   // 8 MB
  float*  lam    = (float*)(ws + 50331648);

  cvt_f32_bf16<<<(4194304 + 255) / 256, 256, 0, stream>>>(x, xb, 4194304);
  cvt_f32_bf16<<<(3145728 + 255) / 256, 256, 0, stream>>>(Wqkv, wqkvb, 3145728);
  cvt_f32_bf16<<<(1048576 + 255) / 256, 256, 0, stream>>>(Wproj, wprojb, 1048576);
  lam_kernel<<<1, 64, 0, stream>>>(lq1, lk1, lq2, lk2, lam);

  qkv_gemm<<<dim3(64, 12), 256, 0, stream>>>(xb, wqkvb, qb, kb, vb);
  // dynamic LDS: 2*64KB scores + 2KB Q + 8KB K/V tile = 141312 B (< 320 KB WGP LDS)
  diff_attn<<<4096, 128, 141312, stream>>>(qb, kb, vb, lam, ob);
  proj_gemm<<<dim3(64, 4), 256, 0, stream>>>(ob, wprojb, bproj, (float*)d_out);
}
